// OuterProductMean_49744311222650
// MI455X (gfx1250) — matrix operand order repaired and checked by reading, not compiled, not run on hardware
//
#include <hip/hip_runtime.h>
#include <stdint.h>

typedef __attribute__((ext_vector_type(16))) __bf16 v16bf;
typedef __attribute__((ext_vector_type(8)))  float  v8f;

#define S_DIM 1024
#define I_DIM 384
#define CM    64
#define C_DIM 32
#define CZ    128
#define MN    (I_DIM * C_DIM)   // 12288
#define NSTEP (S_DIM / 32)      // 32 k-steps of 32

static __device__ __forceinline__ __bf16 to_bf16(float f) {
  union { float f; uint32_t u; } v; v.f = f;
  uint32_t r = v.u + 0x7FFFu + ((v.u >> 16) & 1u);   // round to nearest even
  uint16_t h = (uint16_t)(r >> 16);
  return __builtin_bit_cast(__bf16, h);
}

union Frag { uint4 q[2]; v16bf v; };

// gfx1250 async global->LDS copy, 16B per lane, GVS addressing.
// vdst = per-lane LDS byte address (low 32 bits of flat shared pointer),
// vaddr = per-lane 32-bit byte offset, saddr = uniform 64-bit base.
static __device__ __forceinline__ void async_ld_b128(uint32_t lds_addr,
                                                     const void* sbase,
                                                     uint32_t voff) {
  asm volatile("global_load_async_to_lds_b128 %0, %1, %2"
               :: "v"(lds_addr), "v"(voff), "s"(sbase)
               : "memory");
}

// ---------------------------------------------------------------------------
// Kernel 1: fused LayerNorm + a/b projections + mask, write transposed bf16
//   a_t[(i*32+h)*S + s], b_t[(j*32+h)*S + s]
// ---------------------------------------------------------------------------
__global__ void ln_ab_kernel(const float* __restrict__ x,
                             const float* __restrict__ mask,
                             const float* __restrict__ nw,
                             const float* __restrict__ nb,
                             const float* __restrict__ aw,
                             const float* __restrict__ bw,
                             __bf16* __restrict__ a_t,
                             __bf16* __restrict__ b_t) {
  int g = blockIdx.x * blockDim.x + threadIdx.x;   // 0 .. S*I-1
  int s = g & (S_DIM - 1);
  int i = g >> 10;                                 // S_DIM == 1024
  const float* xr = x + ((size_t)s * I_DIM + i) * CM;

  float m[CM];
  float mu = 0.f;
#pragma unroll
  for (int c = 0; c < CM; c += 4) {
    float4 t = *(const float4*)(xr + c);
    m[c] = t.x; m[c + 1] = t.y; m[c + 2] = t.z; m[c + 3] = t.w;
    mu += t.x + t.y + t.z + t.w;
  }
  mu *= (1.0f / CM);
  float var = 0.f;
#pragma unroll
  for (int c = 0; c < CM; ++c) { float d = m[c] - mu; var += d * d; }
  var *= (1.0f / CM);
  float rs = rsqrtf(var + 1e-5f);
#pragma unroll
  for (int c = 0; c < CM; ++c) m[c] = (m[c] - mu) * rs * nw[c] + nb[c];

  float mk = mask[(size_t)s * I_DIM + i];

  for (int h = 0; h < C_DIM; ++h) {
    const float* awr = aw + h * CM;
    const float* bwr = bw + h * CM;
    float accA = 0.f, accB = 0.f;
#pragma unroll
    for (int c = 0; c < CM; ++c) { accA += m[c] * awr[c]; accB += m[c] * bwr[c]; }
    size_t mrow = (size_t)(i * C_DIM + h);
    a_t[mrow * S_DIM + s] = to_bf16(accA * mk);
    b_t[mrow * S_DIM + s] = to_bf16(accB);
  }
}

// o_weight [Cz][1024] fp32 -> bf16 (already [N][K] layout for WMMA B frags)
__global__ void owconv_kernel(const float* __restrict__ ow,
                              __bf16* __restrict__ owb) {
  int g = blockIdx.x * blockDim.x + threadIdx.x;
  owb[g] = to_bf16(ow[g]);
}

// ---------------------------------------------------------------------------
// Kernel 2: C[12288][12288] = a_t * b_t^T over K=S=1024, 128x128 tile per WG,
// double-buffered async-to-LDS staging, fused [16,1024]x[1024,128] projection.
// 8 waves (4 M x 2 N), each owns 2x4 16x16 WMMA accumulators.
// ---------------------------------------------------------------------------
__global__ void __launch_bounds__(256)
opm_kernel(const __bf16* __restrict__ a_t,
           const __bf16* __restrict__ b_t,
           const __bf16* __restrict__ owb,
           const float*  __restrict__ obias,
           const int*    __restrict__ n_msa,
           float*        __restrict__ out) {
  // two stage buffers: each = A tile [128][40] bf16 (10240B) + B tile (10240B)
  // projection reuses smem: Z = 16 x 1032 bf16 = 33024B <= 40960B
  __shared__ __align__(16) char smem[40960];

  const int tid   = threadIdx.x;
  const int lane  = tid & 31;
  const int wave  = tid >> 5;
  const int wm    = wave >> 1;              // 0..3 -> M offset wm*32
  const int wn    = wave & 1;               // 0..1 -> N offset wn*64
  const int hlane = lane & 15;
  const bool hih  = lane >= 16;
  const int m0 = blockIdx.x * 128;
  const int n0 = blockIdx.y * 128;

  // staging geometry: 128 rows x 64B per tile = 512 x 16B segs, 2 per thread
  const int row0 = tid >> 2,          col0 = (tid & 3) * 8;          // halfs
  const int row1 = (tid + 256) >> 2,  col1 = ((tid + 256) & 3) * 8;
  const uint32_t smem_lds = (uint32_t)(uintptr_t)(char*)smem;
  const uint32_t offA0 = (uint32_t)(row0 * 80 + col0 * 2);
  const uint32_t offA1 = (uint32_t)(row1 * 80 + col1 * 2);
  const uint32_t gA0 = (uint32_t)(((m0 + row0) * S_DIM + col0) * 2);  // bytes
  const uint32_t gA1 = (uint32_t)(((m0 + row1) * S_DIM + col1) * 2);
  const uint32_t gB0 = (uint32_t)(((n0 + row0) * S_DIM + col0) * 2);
  const uint32_t gB1 = (uint32_t)(((n0 + row1) * S_DIM + col1) * 2);

  auto stage = [&](int step, int buf) {
    const uint32_t base = smem_lds + (uint32_t)(buf * 20480);
    const uint32_t kb   = (uint32_t)(step * 64);          // 32 halfs = 64 bytes
    async_ld_b128(base + offA0,         a_t, gA0 + kb);
    async_ld_b128(base + offA1,         a_t, gA1 + kb);
    async_ld_b128(base + 10240 + offA0, b_t, gB0 + kb);
    async_ld_b128(base + 10240 + offA1, b_t, gB1 + kb);
  };

  v8f acc[2][4] = {};

  stage(0, 0);
  stage(1, 1);

  for (int kk = 0; kk < NSTEP; ++kk) {
    const int cur = kk & 1;
    // one 4-instruction batch (step kk+1) may stay in flight; step kk must land
    if (kk + 1 < NSTEP) { asm volatile("s_wait_asynccnt 0x4" ::: "memory"); }
    else                { asm volatile("s_wait_asynccnt 0x0" ::: "memory"); }
    __syncthreads();

    const __bf16* As = (const __bf16*)(smem + cur * 20480);  // [128][40]
    const __bf16* Bs = As + 5120;                            // [128][40]

    // A frag: lane<16 holds K0-7 & K16-23 of row M; lane>=16 K8-15 & K24-31
    Frag af[2];
#pragma unroll
    for (int tm = 0; tm < 2; ++tm) {
      const __bf16* p = As + (wm * 32 + tm * 16 + hlane) * 40 + (hih ? 8 : 0);
      af[tm].q[0] = *(const uint4*)p;
      af[tm].q[1] = *(const uint4*)(p + 16);
    }
    // B frag: lane<16 holds K0-7 & K16-23 of col N; lane>=16 K8-15 & K24-31
    Frag bfr[4];
#pragma unroll
    for (int tn = 0; tn < 4; ++tn) {
      const __bf16* p = Bs + (wn * 64 + tn * 16 + hlane) * 40 + (hih ? 8 : 0);
      bfr[tn].q[0] = *(const uint4*)p;
      bfr[tn].q[1] = *(const uint4*)(p + 16);
    }
#pragma unroll
    for (int tm = 0; tm < 2; ++tm)
#pragma unroll
      for (int tn = 0; tn < 4; ++tn)
        acc[tm][tn] = __builtin_amdgcn_wmma_f32_16x16x32_bf16(
            false, af[tm].v, false, bfr[tn].v, (short)0, acc[tm][tn],
            false, false);

    __syncthreads();
    if (kk + 2 < NSTEP) stage(kk + 2, cur);   // refill buffer just consumed
  }

  // ---- scatter accumulators into Z-LDS as 16 pair-rows of length 1024 ----
  __bf16* Zs = (__bf16*)smem;                 // [16][1032]
  const float inv = 1.0f / (float)n_msa[0];
#pragma unroll
  for (int tm = 0; tm < 2; ++tm)
#pragma unroll
    for (int tn = 0; tn < 4; ++tn)
#pragma unroll
      for (int v = 0; v < 8; ++v) {
        int Ml = wm * 32 + tm * 16 + v + (hih ? 8 : 0);  // local M (i_loc, c)
        int Nl = wn * 64 + tn * 16 + hlane;              // local N (j_loc, d)
        int p  = (Ml >> 5) * 4 + (Nl >> 5);              // pair 0..15
        int kz = (Ml & 31) * 32 + (Nl & 31);             // cd index 0..1023
        Zs[p * 1032 + kz] = to_bf16(acc[tm][tn][v] * inv);
      }
  __syncthreads();

  // ---- projection: [16 pairs x 1024] x o_w^T[1024 x 128], one N-tile/wave ----
  v8f oacc = {};
  const __bf16* owrow =
      owb + (size_t)(wave * 16 + hlane) * 1024 + (hih ? 8 : 0);
  for (int k0 = 0; k0 < 1024; k0 += 32) {
    Frag az, bz;
    const __bf16* pz = Zs + hlane * 1032 + k0 + (hih ? 8 : 0);
    az.q[0] = *(const uint4*)pz;
    az.q[1] = *(const uint4*)(pz + 16);
    bz.q[0] = *(const uint4*)(owrow + k0);
    bz.q[1] = *(const uint4*)(owrow + k0 + 16);
    oacc = __builtin_amdgcn_wmma_f32_16x16x32_bf16(
        false, az.v, false, bz.v, (short)0, oacc, false, false);
  }

  const int zc = wave * 16 + hlane;
  const float bias = obias[zc];
#pragma unroll
  for (int v = 0; v < 8; ++v) {
    int p = v + (hih ? 8 : 0);                 // pair index = projection M row
    int i = blockIdx.x * 4 + (p >> 2);
    int j = blockIdx.y * 4 + (p & 3);
    out[((size_t)i * I_DIM + j) * CZ + zc] = oacc[v] + bias;
  }
}

// ---------------------------------------------------------------------------
extern "C" void kernel_launch(void* const* d_in, const int* in_sizes, int n_in,
                              void* d_out, int out_size, void* d_ws,
                              size_t ws_size, hipStream_t stream) {
  const float* x   = (const float*)d_in[0];
  const float* msk = (const float*)d_in[1];
  const float* nw  = (const float*)d_in[2];
  const float* nb  = (const float*)d_in[3];
  const float* aw  = (const float*)d_in[4];
  const float* bw  = (const float*)d_in[5];
  const float* ow  = (const float*)d_in[6];
  const float* ob  = (const float*)d_in[7];
  const int*   nm  = (const int*)d_in[8];
  float* out = (float*)d_out;

  __bf16* a_t = (__bf16*)d_ws;                     // [12288][1024] bf16
  __bf16* b_t = a_t + (size_t)MN * S_DIM;          // [12288][1024] bf16
  __bf16* owb = b_t + (size_t)MN * S_DIM;          // [128][1024]  bf16

  ln_ab_kernel<<<(S_DIM * I_DIM) / 256, 256, 0, stream>>>(
      x, msk, nw, nb, aw, bw, a_t, b_t);
  owconv_kernel<<<(CZ * 1024) / 256, 256, 0, stream>>>(ow, owb);

  dim3 grid(I_DIM / 4, I_DIM / 4);                 // 96 x 96 tiles of 128x128
  opm_kernel<<<grid, 256, 0, stream>>>(a_t, b_t, owb, ob, nm, out);
}
